// LevitAttention_23665269801101
// MI455X (gfx1250) — matrix operand order repaired and checked by reading, not compiled, not run on hardware
//
#include <hip/hip_runtime.h>
#include <hip/hip_bf16.h>
#include <math.h>

typedef _Float16 half_t;
typedef _Float16 v16h __attribute__((ext_vector_type(16)));
typedef _Float16 v8h  __attribute__((ext_vector_type(8)));
typedef float    v8f  __attribute__((ext_vector_type(8)));
typedef unsigned uint32x4 __attribute__((ext_vector_type(4)));
typedef unsigned uint32x8 __attribute__((ext_vector_type(8)));

#define Bn   256
#define Sn   196
#define Dn   256
#define Hn   8
#define KDn  32
#define VDn  128
#define BSn  (Bn * Sn)          // 50176 rows, divisible by 16
#define QKVn 1536               // H*(32+32+128)
#define PIn  1024               // H*VD
#define SPn  224                // S padded to 14 tiles of 16 (7 K-chunks of 32)
#define QK_PITCH 512            // per-row halfs in qk16 buffer (H * 64)

static __device__ __forceinline__ int imin(int a, int b) { return a < b ? a : b; }

static __device__ __forceinline__ v8f wmma_f16(v16h a, v16h b, v8f c) {
  // D = A(16x32 f16) * B(32x16 f16) + C(16x16 f32)
  return __builtin_amdgcn_wmma_f32_16x16x32_f16(false, a, false, b, (short)0, c, false, false);
}

static __device__ __forceinline__ v16h cat8(v8h lo, v8h hi) {
  return __builtin_shufflevector(lo, hi, 0,1,2,3,4,5,6,7,8,9,10,11,12,13,14,15);
}

// per-lane async memory -> LDS copy, 16 bytes/lane, tracked by ASYNCcnt
static __device__ __forceinline__ void async_copy_b128(unsigned lds_off, const void* g) {
  asm volatile("global_load_async_to_lds_b128 %0, %1, off"
               :: "v"(lds_off), "v"((unsigned long long)(uintptr_t)g)
               : "memory");
}
static __device__ __forceinline__ void wait_asynccnt0() {
  asm volatile("s_wait_asynccnt 0" ::: "memory");
}

// ---------------------------------------------------------------------------
// f32 -> f16 convert
// ---------------------------------------------------------------------------
__global__ __launch_bounds__(256) void cvt_f32_f16(const float* __restrict__ src,
                                                   half_t* __restrict__ dst, int n) {
  for (int i = blockIdx.x * blockDim.x + threadIdx.x; i < n; i += gridDim.x * blockDim.x)
    dst[i] = (half_t)src[i];
}

// ---------------------------------------------------------------------------
// Kernel 1: qkv = BN(hidden @ qkv_w^T), f16 out into qk16 (q|k) and v16 (v)
// grid (3136, 12), block 256 (8 waves). Block: 16 rows x 128 cols.
// A tile staged via GLOBAL_LOAD_ASYNC_TO_LDS_B128 (ASYNCcnt path).
// ---------------------------------------------------------------------------
__global__ __launch_bounds__(256) void qkv_gemm_bn(
    const half_t* __restrict__ hs16,     // (BS, D) f16
    const half_t* __restrict__ w16,      // (QKV, D) f16
    const float* __restrict__ gamma, const float* __restrict__ beta,
    const float* __restrict__ mean,  const float* __restrict__ var,
    half_t* __restrict__ qk16,           // (BS, 512)
    half_t* __restrict__ v16)            // (BS, 1024)
{
  __shared__ half_t As[16][Dn];          // 8 KB
  const int tid = threadIdx.x;
  const int mt  = blockIdx.x;            // M tile (16 rows)
  // async-stage A tile: 256 threads x 32 B = 8 KB, no VGPR data movement
  {
    const int row = tid >> 4;
    const int c0  = (tid & 15) * 16;     // halfs
    const unsigned lds0 = (unsigned)(uintptr_t)&As[row][c0];
    const half_t* g = hs16 + (size_t)(mt * 16 + row) * Dn + c0;
    async_copy_b128(lds0, g);
    async_copy_b128(lds0 + 16, g + 8);
  }
  wait_asynccnt0();
  __syncthreads();

  const int wave = tid >> 5;
  const int lane = tid & 31;
  const int l16  = lane & 15;
  const int hi   = lane >> 4;            // 0 or 1
  const int o    = blockIdx.y * 128 + wave * 16 + l16;  // output channel

  v8f acc = {};
  #pragma unroll
  for (int k0 = 0; k0 < Dn; k0 += 32) {
    v8h a0 = *(const v8h*)&As[l16][k0 + hi * 16];
    v8h a1 = *(const v8h*)&As[l16][k0 + hi * 16 + 8];
    v16h a = cat8(a0, a1);
    v16h b = *(const v16h*)&w16[(size_t)o * Dn + k0 + hi * 16];
    acc = wmma_f16(a, b, acc);
  }

  const float sc = gamma[o] * rsqrtf(var[o] + 1e-5f);
  const float sh = beta[o] - mean[o] * sc;
  const int hh = o / 192, rr = o % 192;
  #pragma unroll
  for (int r = 0; r < 8; ++r) {
    const int m = r + hi * 8;
    const size_t row_g = (size_t)(mt * 16 + m);
    const float val = acc[r] * sc + sh;
    if (rr < 64) qk16[row_g * QK_PITCH + hh * 64 + rr] = (half_t)val;
    else         v16 [row_g * PIn      + hh * 128 + (rr - 64)] = (half_t)val;
  }
}

// ---------------------------------------------------------------------------
// Kernel 2: attention per (b,h). One wave per workgroup.
// logits in 112 VGPRs; softmax via width-16 shfl_xor; PV via WMMA with vT in
// LDS. Output = hardswish(attn@v) overwrites the v16 slice in place.
// ---------------------------------------------------------------------------
__global__ __launch_bounds__(32) void attention_kernel(
    const half_t* __restrict__ qk16,     // (BS, 512)
    half_t* __restrict__ v16,            // (BS, 1024), overwritten in place
    const float* __restrict__ biases,    // (H, n_off)
    const int* __restrict__ bidx,        // (S, S)
    int n_off)
{
  __shared__ half_t vT[VDn][SPn];        // 57344 B
  __shared__ half_t attn_s[16][SPn];     //  7168 B
  const int b = blockIdx.x, h = blockIdx.y;
  const int lane = threadIdx.x;
  const int l16  = lane & 15;
  const int hi   = lane >> 4;
  const float scale = 0.1767766953f;     // 32^-0.5

  // stage v^T: vT[d][s] = v16[b, s, h, d]  (zero pad s >= 196)
  for (int i = lane; i < SPn * (VDn / 2); i += 32) {
    const int dp = (i & 63) * 2;         // d pair (coalesced across lanes)
    const int s  = i >> 6;
    half_t a0 = (half_t)0.0f, a1 = (half_t)0.0f;
    if (s < Sn) {
      const half_t* vr = v16 + (size_t)(b * Sn + s) * PIn + h * VDn + dp;
      a0 = vr[0]; a1 = vr[1];
    }
    vT[dp][s] = a0;
    vT[dp + 1][s] = a1;
  }
  __syncthreads();

  for (int qt = 0; qt < SPn / 16; ++qt) {
    // prefetch next q-tile rows (global_prefetch_b8)
    if (qt + 1 < SPn / 16) {
      const int pr = imin((qt + 1) * 16 + l16, Sn - 1);
      __builtin_prefetch(qk16 + (size_t)(b * Sn + pr) * QK_PITCH + h * 64, 0, 3);
    }

    // ---- Q*K^T logits ----
    const int qrow = imin(qt * 16 + l16, Sn - 1);
    const half_t* qp = qk16 + (size_t)(b * Sn + qrow) * QK_PITCH + h * 64 + hi * 16;
    v16h qa = cat8(*(const v8h*)qp, *(const v8h*)(qp + 8));

    v8f lg[14];
    #pragma unroll
    for (int kt = 0; kt < 14; ++kt) {
      const int krow = imin(kt * 16 + l16, Sn - 1);
      v16h kb = *(const v16h*)(qk16 + (size_t)(b * Sn + krow) * QK_PITCH + h * 64 + 32 + hi * 16);
      v8f z = {};
      lg[kt] = wmma_f16(qa, kb, z);
    }

    // ---- scale + relative-position bias + mask ----
    #pragma unroll
    for (int kt = 0; kt < 14; ++kt) {
      const int ka = kt * 16 + l16;
      #pragma unroll
      for (int r = 0; r < 8; ++r) {
        const int qa_i = imin(qt * 16 + r + hi * 8, Sn - 1);
        float x;
        if (ka < Sn)
          x = lg[kt][r] * scale + biases[h * n_off + bidx[qa_i * Sn + ka]];
        else
          x = -1e30f;
        lg[kt][r] = x;
      }
    }

    // ---- softmax per row (rows r and r+8 live in the two lane-halves) ----
    #pragma unroll
    for (int r = 0; r < 8; ++r) {
      float mx = lg[0][r];
      #pragma unroll
      for (int kt = 1; kt < 14; ++kt) mx = fmaxf(mx, lg[kt][r]);
      #pragma unroll
      for (int d = 8; d >= 1; d >>= 1) mx = fmaxf(mx, __shfl_xor(mx, d, 16));
      float sm = 0.f;
      #pragma unroll
      for (int kt = 0; kt < 14; ++kt) {
        float e = __expf(lg[kt][r] - mx);
        lg[kt][r] = e;
        sm += e;
      }
      #pragma unroll
      for (int d = 8; d >= 1; d >>= 1) sm += __shfl_xor(sm, d, 16);
      const float inv = 1.0f / sm;
      const int m = r + hi * 8;
      #pragma unroll
      for (int kt = 0; kt < 14; ++kt)
        attn_s[m][kt * 16 + l16] = (half_t)(lg[kt][r] * inv);
    }
    __syncthreads();

    // ---- out = attn @ v  (16 x 128) ----
    v8f oac[8];
    #pragma unroll
    for (int nt = 0; nt < 8; ++nt) oac[nt] = (v8f){};
    #pragma unroll
    for (int kk = 0; kk < SPn / 32; ++kk) {
      const half_t* ap = &attn_s[l16][kk * 32 + hi * 16];
      v16h aa = cat8(*(const v8h*)ap, *(const v8h*)(ap + 8));
      #pragma unroll
      for (int nt = 0; nt < 8; ++nt) {
        v16h bb = *(const v16h*)&vT[nt * 16 + l16][kk * 32 + hi * 16];
        oac[nt] = wmma_f16(aa, bb, oac[nt]);
      }
    }

    // ---- hardswish + in-place store over the v slice ----
    #pragma unroll
    for (int nt = 0; nt < 8; ++nt) {
      #pragma unroll
      for (int r = 0; r < 8; ++r) {
        const int qa_i = qt * 16 + r + hi * 8;
        if (qa_i < Sn) {
          float x = oac[nt][r];
          float hsw = x * fminf(fmaxf(x + 3.0f, 0.0f), 6.0f) * (1.0f / 6.0f);
          v16[(size_t)(b * Sn + qa_i) * PIn + h * VDn + nt * 16 + l16] = (half_t)hsw;
        }
      }
    }
    __syncthreads();
  }
}

// ---------------------------------------------------------------------------
// Kernel 3: out = BN(hardswish_out @ proj_w^T), f32 output.
// grid (3136, 2), block 256 (8 waves). Block: 16 rows x 128 cols.
// A tile (16 x 2048 B) staged with the Tensor Data Mover (TENSORcnt path).
// ---------------------------------------------------------------------------
__global__ __launch_bounds__(256) void proj_gemm_bn(
    const half_t* __restrict__ a16,      // (BS, 1024) f16 (hardswished)
    const half_t* __restrict__ w16,      // (D, 1024) f16
    const float* __restrict__ gamma, const float* __restrict__ beta,
    const float* __restrict__ mean,  const float* __restrict__ var,
    float* __restrict__ out)             // (BS, D) f32
{
  __shared__ half_t As[16][PIn];         // 32 KB
  const int tid = threadIdx.x;
  const int mt  = blockIdx.x;

  // TDM: 2D tile, 16 rows x 1024 f16 elements, row-major -> contiguous LDS.
  if (tid < 32) {                        // wave 0 issues the DMA
    const unsigned lds = (unsigned)(uintptr_t)&As[0][0];
    const unsigned long long ga =
        (unsigned long long)(uintptr_t)(a16 + (size_t)mt * 16 * PIn);
    // D# group 0: count=1 | lds_addr | global_addr[56:0] | type=2
    uint32x4 g0 = { 1u, lds, (unsigned)ga,
                    (unsigned)((ga >> 32) & 0x01FFFFFFu) | 0x80000000u };
    // D# group 1: data_size=2B; tensor_dim0=1024; tensor_dim1=BS;
    //             tile_dim0=1024; tile_dim1=16; tensor_dim0_stride=1024
    uint32x8 g1 = { 0x00010000u,
                    (unsigned)((PIn & 0xFFFFu) << 16),
                    ((unsigned)(BSn & 0xFFFFu) << 16) | ((PIn >> 16) & 0xFFFFu),
                    ((unsigned)(PIn & 0xFFFFu) << 16) | ((BSn >> 16) & 0xFFFFu),
                    16u,
                    (unsigned)PIn,
                    0u, 0u };
    uint32x4 gz = { 0u, 0u, 0u, 0u };    // groups 2/3: dims/tiles unused (zero)
    asm volatile("tensor_load_to_lds %0, %1, %2, %3"
                 :: "s"(g0), "s"(g1), "s"(gz), "s"(gz) : "memory");
    __builtin_amdgcn_s_wait_tensorcnt(0);
  }
  __syncthreads();

  const int wave = tid >> 5;
  const int lane = tid & 31;
  const int l16  = lane & 15;
  const int hi   = lane >> 4;
  const int o    = blockIdx.y * 128 + wave * 16 + l16;

  v8f acc = {};
  #pragma unroll 8
  for (int k0 = 0; k0 < PIn; k0 += 32) {
    v8h a0 = *(const v8h*)&As[l16][k0 + hi * 16];
    v8h a1 = *(const v8h*)&As[l16][k0 + hi * 16 + 8];
    v16h a = cat8(a0, a1);
    v16h b = *(const v16h*)&w16[(size_t)o * PIn + k0 + hi * 16];
    acc = wmma_f16(a, b, acc);
  }

  const float sc = gamma[o] * rsqrtf(var[o] + 1e-5f);
  const float sh = beta[o] - mean[o] * sc;
  #pragma unroll
  for (int r = 0; r < 8; ++r) {
    const int m = r + hi * 8;
    out[(size_t)(mt * 16 + m) * Dn + o] = acc[r] * sc + sh;
  }
}

// ---------------------------------------------------------------------------
extern "C" void kernel_launch(void* const* d_in, const int* in_sizes, int n_in,
                              void* d_out, int out_size, void* d_ws, size_t ws_size,
                              hipStream_t stream) {
  const float* hs         = (const float*)d_in[0];
  const float* qkv_w      = (const float*)d_in[1];
  const float* qkv_gamma  = (const float*)d_in[2];
  const float* qkv_beta   = (const float*)d_in[3];
  const float* qkv_mean   = (const float*)d_in[4];
  const float* qkv_var    = (const float*)d_in[5];
  const float* proj_w     = (const float*)d_in[6];
  const float* proj_gamma = (const float*)d_in[7];
  const float* proj_beta  = (const float*)d_in[8];
  const float* proj_mean  = (const float*)d_in[9];
  const float* proj_var   = (const float*)d_in[10];
  const float* biases     = (const float*)d_in[11];
  const int*   bidx       = (const int*)d_in[12];
  const int n_off = in_sizes[11] / Hn;

  // workspace carve (all offsets 32B-aligned)
  half_t* qkvw16  = (half_t*)d_ws;                     // 1536*256
  half_t* projw16 = qkvw16 + (size_t)QKVn * Dn;        // 256*1024
  half_t* hs16    = projw16 + (size_t)Dn * PIn;        // BS*256
  half_t* qk16    = hs16 + (size_t)BSn * Dn;           // BS*512
  half_t* v16     = qk16 + (size_t)BSn * QK_PITCH;     // BS*1024

  cvt_f32_f16<<<256, 256, 0, stream>>>(qkv_w, qkvw16, QKVn * Dn);
  cvt_f32_f16<<<256, 256, 0, stream>>>(proj_w, projw16, Dn * PIn);
  cvt_f32_f16<<<4096, 256, 0, stream>>>(hs, hs16, BSn * Dn);

  qkv_gemm_bn<<<dim3(BSn / 16, QKVn / 128), 256, 0, stream>>>(
      hs16, qkvw16, qkv_gamma, qkv_beta, qkv_mean, qkv_var, qk16, v16);

  attention_kernel<<<dim3(Bn, Hn), 32, 0, stream>>>(qk16, v16, biases, bidx, n_off);

  proj_gemm_bn<<<dim3(BSn / 16, Dn / 128), 256, 0, stream>>>(
      v16, projw16, proj_gamma, proj_beta, proj_mean, proj_var, (float*)d_out);
}
